// ActorCritic_88545045774843
// MI455X (gfx1250) — matrix operand order repaired and checked by reading, not compiled, not run on hardware
//
#include <hip/hip_runtime.h>
#include <hip/hip_bf16.h>

// ---------------------------------------------------------------------------
// Types for CDNA5 WMMA (wave32): v_wmma_f32_16x16x32_bf16
// ---------------------------------------------------------------------------
typedef __bf16 bf16_t;
typedef __attribute__((ext_vector_type(16))) __bf16 v16bf;
typedef __attribute__((ext_vector_type(8)))  float  v8f;

#define NN   20000
#define BB   8
#define RR   160000      // N*B rows
#define HH   1024
#define EE   320000
#define KP0  64          // IN=42 padded to 64 (K multiple of 32)
#define OUTN 21
#define MAXN 20000.0f

// workspace layout (bytes, all 256-aligned)
static constexpr size_t OFF_HF32 = 0;                                   // [RR][64] f32
static constexpr size_t OFF_HBF  = OFF_HF32 + (size_t)RR * KP0 * 4;     // [RR][64] bf16
static constexpr size_t OFF_W0A  = OFF_HBF  + (size_t)RR * KP0 * 2;     // [1024][64] bf16 (aW0^T)
static constexpr size_t OFF_W1A  = OFF_W0A  + (size_t)HH * KP0 * 2;     // [1024][1024] bf16 (aW1^T)
static constexpr size_t OFF_W2A  = OFF_W1A  + (size_t)HH * HH  * 2;     // [32][1024] bf16 (aW2^T pad)
static constexpr size_t OFF_W0C  = OFF_W2A  + (size_t)32 * HH  * 2;     // [1024][64] bf16 (cW0^T)
static constexpr size_t OFF_W1C  = OFF_W0C  + (size_t)HH * KP0 * 2;     // [1024][1024] bf16 (cW1^T)
static constexpr size_t OFF_SCAL = OFF_W1C  + (size_t)HH * HH  * 2;     // ent_sum, mask_sum

// dynamic LDS per WG: x1 (4 waves * 16x1024 bf16) = 128KB
//                   + x2 chunk stage (4 * 16x64 bf16) = 8KB
//                   + logits (4 * 16x32 f32) = 8KB          => 144KB, 2 WGs/WGP
#define X1_BYTES  (4 * 16 * HH * 2)
#define X2C_BYTES (4 * 16 * 64 * 2)
#define SMEM_BYTES (X1_BYTES + X2C_BYTES + 4 * 16 * 32 * 4)   // 147456

// ---------------------------------------------------------------------------
// fragment loaders (layouts per CDNA5 ISA 7.12.2)
// A-frag (16-bit A 16x32): lane m = lane&15; K groups {g..g+7, g+16..g+23}, g=(lane>>4)*8
// B-frag (32x16): lane n = lane&15; K groups {g..g+7, g+16..g+23}, g=(lane>>4)*8
// ---------------------------------------------------------------------------
__device__ inline v16bf ldA_frag(const bf16_t* p) {     // p[0..7], p[16..23]
    union { uint4 u[2]; v16bf v; } t;
    t.u[0] = *reinterpret_cast<const uint4*>(p);
    t.u[1] = *reinterpret_cast<const uint4*>(p + 16);
    return t.v;
}
__device__ inline v16bf ldB_frag(const bf16_t* p) {     // p[0..7], p[16..23]
    union { uint4 u[2]; v16bf v; } t;
    t.u[0] = *reinterpret_cast<const uint4*>(p);
    t.u[1] = *reinterpret_cast<const uint4*>(p + 16);
    return t.v;
}
__device__ inline v8f wmma_bf16(v16bf a, v16bf b, v8f c) {
    return __builtin_amdgcn_wmma_f32_16x16x32_bf16(false, a, false, b, (short)0, c,
                                                   false, false);
}

// ---------------------------------------------------------------------------
// prep kernels
// ---------------------------------------------------------------------------
__global__ void zero_small(float* scal, float* out_val) {
    int i = threadIdx.x;
    if (i < 2) scal[i] = 0.0f;
    if (i < BB) out_val[i] = 0.0f;
}

__global__ void init_h(const float* __restrict__ obt, float* __restrict__ h) {
    int r = blockIdx.x * blockDim.x + threadIdx.x;
    if (r >= RR) return;
    float* p = h + (size_t)r * KP0;
#pragma unroll
    for (int i = 0; i < KP0; ++i) p[i] = 0.0f;
    p[0]  = obt[r];
    p[41] = 1.0f;
}

__global__ void edge_scatter(const int* __restrict__ src, const int* __restrict__ dst,
                             const int* __restrict__ obx, float* __restrict__ h) {
    int i = blockIdx.x * blockDim.x + threadIdx.x;
    if (i >= EE * BB) return;
    int e = i >> 3, b = i & 7;
    int s = src[e], d = dst[e];
    int cs = obx[s * BB + b];                       // forward: onehot(src) summed at dst
    if (cs > 0) atomicAdd(h + ((size_t)d * BB + b) * KP0 + cs, 1.0f);
    int cd = obx[d * BB + b];                       // reverse: onehot(dst) summed at src
    if (cd > 0) atomicAdd(h + ((size_t)s * BB + b) * KP0 + 20 + cd, 1.0f);
}

__global__ void cvt_h(const float* __restrict__ src, bf16_t* __restrict__ dst) {
    int i = blockIdx.x * blockDim.x + threadIdx.x;   // one thread per 4 elems
    if (i >= RR * KP0 / 4) return;
    float4 v = reinterpret_cast<const float4*>(src)[i];
    union { bf16_t h[4]; uint2 u; } t;
    t.h[0] = (bf16_t)v.x; t.h[1] = (bf16_t)v.y;
    t.h[2] = (bf16_t)v.z; t.h[3] = (bf16_t)v.w;
    reinterpret_cast<uint2*>(dst)[i] = t.u;
}

// W [Ksrc][Nsrc] f32  ->  Wt [Npad][Kpad] bf16 (transposed, zero padded)
__global__ void wcvt(const float* __restrict__ W, bf16_t* __restrict__ Wt,
                     int Ksrc, int Nsrc, int Kpad, int total) {
    int i = blockIdx.x * blockDim.x + threadIdx.x;
    if (i >= total) return;
    int n = i / Kpad, k = i % Kpad;
    float v = (k < Ksrc && n < Nsrc) ? W[(size_t)k * Nsrc + n] : 0.0f;
    Wt[i] = (bf16_t)v;
}

// ---------------------------------------------------------------------------
// fused 3-layer MLP: 64 rows per WG, 16 rows per wave.
// Layer 3 is fused into the layer-2 N-chunk loop so x2 never fully materializes:
//  ACTOR : stage 16x64 chunk in LDS -> 2 layer-3 WMMA k-steps, logits accumulate
//  CRITIC: per-lane register dot-product accumulation (no LDS, f32 precision)
// ---------------------------------------------------------------------------
template <bool ACTOR>
__global__ __launch_bounds__(128, 2)
void mlp_fused(const bf16_t* __restrict__ hbf,
               const bf16_t* __restrict__ W0t, const float* __restrict__ b0,
               const bf16_t* __restrict__ W1t, const float* __restrict__ b1,
               const bf16_t* __restrict__ W2t, const float* __restrict__ b2,
               const float*  __restrict__ cw2, const float* __restrict__ cb2,
               const int* __restrict__ obx, const int* __restrict__ action,
               float* __restrict__ out_alp, float* __restrict__ out_val,
               float* __restrict__ scal) {
    extern __shared__ char smem_raw[];
    const int lane  = threadIdx.x & 31;
    const int wave  = threadIdx.x >> 5;
    const int lmod  = lane & 15;
    const int lhalf = lane >> 4;
    const int kgA   = lhalf * 8;     // A-frag K group offset
    const int kgB   = lhalf * 8;     // B-frag K group offset
    const int rowbase = blockIdx.x * 64 + wave * 16;

    bf16_t* x1  = reinterpret_cast<bf16_t*>(smem_raw) + (size_t)wave * 16 * HH;
    bf16_t* x2c = reinterpret_cast<bf16_t*>(smem_raw + X1_BYTES) + (size_t)wave * 16 * 64;
    float*  lg  = reinterpret_cast<float*>(smem_raw + X1_BYTES + X2C_BYTES) + wave * 16 * 32;

    // ---------------- layer 1: [16 x 64] x [64 x 1024] ----------------
    const bf16_t* hrow = hbf + (size_t)(rowbase + lmod) * KP0;
    v16bf a0 = ldA_frag(hrow + 0  + kgA);
    v16bf a1 = ldA_frag(hrow + 32 + kgA);

    for (int c = 0; c < 8; ++c) {            // 8 chunks x 8 N-blocks = 1024 cols
        v8f acc[8];
#pragma unroll
        for (int t = 0; t < 8; ++t) {
            int n = (c * 8 + t) * 16 + lmod;
            float bv = b0[n];
#pragma unroll
            for (int g = 0; g < 8; ++g) acc[t][g] = bv;
            const bf16_t* bp = W0t + (size_t)n * KP0;
            acc[t] = wmma_bf16(a0, ldB_frag(bp + kgB), acc[t]);
            acc[t] = wmma_bf16(a1, ldB_frag(bp + 32 + kgB), acc[t]);
        }
#pragma unroll
        for (int t = 0; t < 8; ++t) {
            int n = (c * 8 + t) * 16 + lmod;
#pragma unroll
            for (int g = 0; g < 8; ++g) {
                float v = acc[t][g];
                x1[(g + lhalf * 8) * HH + n] = (bf16_t)(v > 0.0f ? v : 0.0f);
            }
        }
    }

    // persistent layer-3 accumulators
    v8f   acc3[2];                  // actor logits (cols 0..31)
    float sacc[8];                  // critic row partial dots
    if (ACTOR) {
#pragma unroll
        for (int t = 0; t < 2; ++t) {
            int n = t * 16 + lmod;
            float bv = (n < OUTN) ? b2[n] : 0.0f;
#pragma unroll
            for (int g = 0; g < 8; ++g) acc3[t][g] = bv;
        }
    } else {
#pragma unroll
        for (int g = 0; g < 8; ++g) sacc[g] = 0.0f;
    }

    // -------- layer 2 (16 chunks of 64 cols) with fused layer 3 --------
    for (int c = 0; c < 16; ++c) {
        v8f acc[4];
#pragma unroll
        for (int t = 0; t < 4; ++t) {
            float bv = b1[(c * 4 + t) * 16 + lmod];
#pragma unroll
            for (int g = 0; g < 8; ++g) acc[t][g] = bv;
        }
        for (int ks = 0; ks < 32; ++ks) {
            v16bf a = ldA_frag(x1 + lmod * HH + ks * 32 + kgA);
#pragma unroll
            for (int t = 0; t < 4; ++t) {
                int n = (c * 4 + t) * 16 + lmod;
                acc[t] = wmma_bf16(a, ldB_frag(W1t + (size_t)n * HH + ks * 32 + kgB),
                                   acc[t]);
            }
        }

        if (ACTOR) {
            // stage relu(chunk) as bf16 16x64 tile, then 2 layer-3 k-steps
#pragma unroll
            for (int t = 0; t < 4; ++t) {
                int nl = t * 16 + lmod;
#pragma unroll
                for (int g = 0; g < 8; ++g) {
                    float v = acc[t][g];
                    x2c[(g + lhalf * 8) * 64 + nl] = (bf16_t)(v > 0.0f ? v : 0.0f);
                }
            }
#pragma unroll
            for (int j = 0; j < 2; ++j) {
                v16bf a = ldA_frag(x2c + lmod * 64 + j * 32 + kgA);
#pragma unroll
                for (int t = 0; t < 2; ++t) {
                    int n = t * 16 + lmod;
                    acc3[t] = wmma_bf16(
                        a, ldB_frag(W2t + (size_t)n * HH + (c * 2 + j) * 32 + kgB),
                        acc3[t]);
                }
            }
        } else {
            // critic: accumulate dot(relu(x2_row), cW2) directly in registers
#pragma unroll
            for (int t = 0; t < 4; ++t) {
                int n = (c * 4 + t) * 16 + lmod;
                float w = cw2[n];
#pragma unroll
                for (int g = 0; g < 8; ++g) {
                    float v = acc[t][g];
                    sacc[g] += (v > 0.0f ? v : 0.0f) * w;
                }
            }
        }
    }

    if (ACTOR) {
        // ---- epilogue: log-softmax, logp[action]*mask, entropy ----
#pragma unroll
        for (int t = 0; t < 2; ++t)
#pragma unroll
            for (int g = 0; g < 8; ++g)
                lg[(g + lhalf * 8) * 32 + t * 16 + lmod] = acc3[t][g];

        float entv = 0.0f, mv = 0.0f;
        if (lane < 16) {
            int r = rowbase + lane;
            const float* L = lg + lane * 32;
            float mx = L[0];
            for (int cc = 1; cc < OUTN; ++cc) mx = fmaxf(mx, L[cc]);
            float s = 0.0f;
            for (int cc = 0; cc < OUTN; ++cc) s += __expf(L[cc] - mx);
            float lse = __logf(s) + mx;
            float msk = (obx[r] == 0) ? 1.0f : 0.0f;
            out_alp[r] = (L[action[r]] - lse) * msk;
            float ent = 0.0f;
            for (int cc = 0; cc < OUTN; ++cc) {
                float p = __expf(L[cc] - lse);
                ent += p * (lse - L[cc]);
            }
            entv = ent * msk;
            mv   = msk;
        }
        for (int off = 16; off >= 1; off >>= 1) {
            entv += __shfl_xor(entv, off);
            mv   += __shfl_xor(mv, off);
        }
        if (lane == 0) {
            atomicAdd(&scal[0], entv);
            atomicAdd(&scal[1], mv);
        }
    } else {
        // ---- critic epilogue: 16-lane column reduction, 8 atomics per half ----
#pragma unroll
        for (int g = 0; g < 8; ++g) {
            float s = sacc[g];
            s += __shfl_xor(s, 1);
            s += __shfl_xor(s, 2);
            s += __shfl_xor(s, 4);
            s += __shfl_xor(s, 8);
            sacc[g] = s;                 // complete in lanes with lmod==0
        }
        if (lmod == 0) {
            float bv = cb2[0];
#pragma unroll
            for (int g = 0; g < 8; ++g) {
                int m = g + lhalf * 8;           // row in tile
                int r = rowbase + m;
                float msk = (obx[r] == 0) ? 1.0f : 0.0f;
                // rowbase % 8 == 0  =>  batch index b = m % 8 = g
                atomicAdd(&out_val[g], (sacc[g] + bv) * msk * (1.0f / MAXN));
            }
        }
    }
}

__global__ void finalize(const float* scal, float* out_ent) {
    out_ent[0] = scal[0] / fmaxf(scal[1], 1.0f);
}

// ---------------------------------------------------------------------------
extern "C" void kernel_launch(void* const* d_in, const int* in_sizes, int n_in,
                              void* d_out, int out_size, void* d_ws, size_t ws_size,
                              hipStream_t stream) {
    const int*   obx    = (const int*)d_in[0];
    const float* obt    = (const float*)d_in[1];
    const int*   action = (const int*)d_in[2];
    const int*   src    = (const int*)d_in[3];
    const int*   dst    = (const int*)d_in[4];
    const float* aW0 = (const float*)d_in[5],  *ab0 = (const float*)d_in[6];
    const float* aW1 = (const float*)d_in[7],  *ab1 = (const float*)d_in[8];
    const float* aW2 = (const float*)d_in[9],  *ab2 = (const float*)d_in[10];
    const float* cW0 = (const float*)d_in[11], *cb0 = (const float*)d_in[12];
    const float* cW1 = (const float*)d_in[13], *cb1 = (const float*)d_in[14];
    const float* cW2 = (const float*)d_in[15], *cb2 = (const float*)d_in[16];

    char* ws = (char*)d_ws;
    float*  hf32 = (float*)(ws + OFF_HF32);
    bf16_t* hbf  = (bf16_t*)(ws + OFF_HBF);
    bf16_t* w0a  = (bf16_t*)(ws + OFF_W0A);
    bf16_t* w1a  = (bf16_t*)(ws + OFF_W1A);
    bf16_t* w2a  = (bf16_t*)(ws + OFF_W2A);
    bf16_t* w0c  = (bf16_t*)(ws + OFF_W0C);
    bf16_t* w1c  = (bf16_t*)(ws + OFF_W1C);
    float*  scal = (float*)(ws + OFF_SCAL);

    float* out     = (float*)d_out;
    float* out_alp = out;             // [N,B] flat, row r = n*B+b
    float* out_ent = out + RR;        // scalar
    float* out_val = out + RR + 1;    // [B]

    hipFuncSetAttribute((const void*)mlp_fused<true>,
                        hipFuncAttributeMaxDynamicSharedMemorySize, SMEM_BYTES);
    hipFuncSetAttribute((const void*)mlp_fused<false>,
                        hipFuncAttributeMaxDynamicSharedMemorySize, SMEM_BYTES);

    zero_small<<<1, 32, 0, stream>>>(scal, out_val);
    init_h<<<(RR + 255) / 256, 256, 0, stream>>>(obt, hf32);
    edge_scatter<<<(EE * BB + 255) / 256, 256, 0, stream>>>(src, dst, obx, hf32);
    cvt_h<<<(RR * KP0 / 4 + 255) / 256, 256, 0, stream>>>(hf32, hbf);

    wcvt<<<(HH * KP0 + 255) / 256, 256, 0, stream>>>(aW0, w0a, 42, HH, KP0, HH * KP0);
    wcvt<<<(HH * HH + 255) / 256, 256, 0, stream>>>(aW1, w1a, HH, HH, HH, HH * HH);
    wcvt<<<(32 * HH + 255) / 256, 256, 0, stream>>>(aW2, w2a, HH, OUTN, HH, 32 * HH);
    wcvt<<<(HH * KP0 + 255) / 256, 256, 0, stream>>>(cW0, w0c, 42, HH, KP0, HH * KP0);
    wcvt<<<(HH * HH + 255) / 256, 256, 0, stream>>>(cW1, w1c, HH, HH, HH, HH * HH);

    mlp_fused<true><<<RR / 64, 128, SMEM_BYTES, stream>>>(
        hbf, w0a, ab0, w1a, ab1, w2a, ab2, cW2, cb2,
        obx, action, out_alp, out_val, scal);
    mlp_fused<false><<<RR / 64, 128, SMEM_BYTES, stream>>>(
        hbf, w0c, cb0, w1c, cb1, w2a /*unused*/, cb0 /*unused*/, cW2, cb2,
        obx, action, out_alp, out_val, scal);

    finalize<<<1, 1, 0, stream>>>(scal, out_ent);
}
